// ADMM_Layer_7902739824983
// MI455X (gfx1250) — matrix operand order repaired and checked by reading, not compiled, not run on hardware
//
#include <hip/hip_runtime.h>
#include <hip/hip_bf16.h>

typedef __attribute__((ext_vector_type(16))) _Float16 v16h;
typedef __attribute__((ext_vector_type(8)))  float    v8f;

// ---------------------------------------------------------------------------
// Kernel 0: zero workspace (atomics accumulate -> must re-zero every launch)
// ---------------------------------------------------------------------------
__global__ void zero_ws_kernel(float* __restrict__ ws, long n) {
    long i = (long)blockIdx.x * blockDim.x + threadIdx.x;
    if (i < n) ws[i] = 0.0f;
}

// ---------------------------------------------------------------------------
// Kernel 1: edge pass 1 — weighted gather from senders, atomic scatter to
// receivers: lam_agg, y_agg, w_deg, deg_sq.  Streaming, atomic-bound.
// ---------------------------------------------------------------------------
__global__ void edge_pass1_kernel(const float* __restrict__ edges,
                                  const int*   __restrict__ senders,
                                  const int*   __restrict__ receivers,
                                  const float* __restrict__ lam,
                                  const float* __restrict__ y,
                                  float* __restrict__ lam_agg,
                                  float* __restrict__ y_agg,
                                  float* __restrict__ w_deg,
                                  float* __restrict__ deg_sq,
                                  long E) {
    long stride = (long)gridDim.x * blockDim.x;
    for (long e = (long)blockIdx.x * blockDim.x + threadIdx.x; e < E; e += stride) {
        if (e + stride < E) {  // speculative prefetch one stride ahead
            __builtin_prefetch(&edges[e + stride], 0, 0);
            __builtin_prefetch(&senders[e + stride], 0, 0);
            __builtin_prefetch(&receivers[e + stride], 0, 0);
        }
        float w = edges[e];
        int s = senders[e];
        int r = receivers[e];
        float l0 = lam[2*s], l1 = lam[2*s + 1];
        float q0 = y[2*s],   q1 = y[2*s + 1];
        atomicAdd(&lam_agg[2*r],     -w * l0);
        atomicAdd(&lam_agg[2*r + 1], -w * l1);
        atomicAdd(&y_agg[2*r],       -w * q0);
        atomicAdd(&y_agg[2*r + 1],   -w * q1);
        atomicAdd(&w_deg[r],  w);
        atomicAdd(&deg_sq[r], w * w);
    }
}

// ---------------------------------------------------------------------------
// Kernel 2: per-node MLP [11 -> 32 -> 1] with per-sample normalization.
// Layer 1 runs on v_wmma_f32_16x16x32_f16: each wave owns 32 nodes, stages a
// 32x32 f16 A-tile (K padded 11->32) in LDS, and does 2 M-tiles x 2 N-tiles.
// Layer 2 (32->1) is a per-lane dot + softplus; alpha reduced to one scalar.
// EXEC stays all-ones through the WMMA region (tail nodes clamped, only the
// final atomicAdd is guarded).
// ---------------------------------------------------------------------------
__global__ __launch_bounds__(128)
void node_mlp_alpha_kernel(const float* __restrict__ x,
                           const float* __restrict__ y,
                           const float* __restrict__ lam,
                           const float* __restrict__ lam_agg,
                           const float* __restrict__ y_agg,
                           const float* __restrict__ w_deg,
                           const float* __restrict__ W1,   // [11,32] row-major
                           const float* __restrict__ b1,   // [32]
                           const float* __restrict__ W2,   // [32]
                           const float* __restrict__ b2,   // [1]
                           float* __restrict__ alpha_sum,
                           long N) {
    __shared__ _Float16 W1s[32][32];        // K-padded weights, 2 KB
    __shared__ _Float16 Alds[4][32][32];    // per-wave A tiles, 8 KB
    __shared__ float    Hlds[4][32][32];    // per-wave hidden acts, 16 KB

    const int tid  = threadIdx.x;
    const int wave = tid >> 5;
    const int lane = tid & 31;
    const int hs   = lane >> 4;     // half-select (lanes 16-31)
    const int lrow = lane & 15;

    // Stage W1 (padded K: rows 11..31 = 0)
    for (int i = tid; i < 32 * 32; i += 128) {
        int k = i >> 5, nn = i & 31;
        W1s[k][nn] = (k < 11) ? (_Float16)W1[k * 32 + nn] : (_Float16)0.0f;
    }

    const long nodeBase = (long)blockIdx.x * 128 + (long)wave * 32;
    const long node = nodeBase + lane;
    const long ni = (node < N) ? node : (N - 1);   // clamp tail, keep EXEC full

    // Build the 11-feature input: [x0,x1, y0,y1, yagg0,yagg1, l0,l1, lagg0,lagg1, wdeg]
    float inp[11];
    inp[0] = x[2*ni];       inp[1] = x[2*ni + 1];
    inp[2] = y[2*ni];       inp[3] = y[2*ni + 1];
    inp[4] = y_agg[2*ni];   inp[5] = y_agg[2*ni + 1];
    inp[6] = lam[2*ni];     inp[7] = lam[2*ni + 1];
    inp[8] = lam_agg[2*ni]; inp[9] = lam_agg[2*ni + 1];
    inp[10] = w_deg[ni];

    float mu = 0.0f;
    #pragma unroll
    for (int k = 0; k < 11; ++k) mu += inp[k];
    mu *= (1.0f / 11.0f);
    float var = 0.0f;
    #pragma unroll
    for (int k = 0; k < 11; ++k) { float d = inp[k] - mu; var += d * d; }
    var *= (1.0f / 11.0f);
    float inv = 1.0f / (sqrtf(var) + 1e-8f);

    // Lane writes its own node row (K padded to 32) into the A tile
    #pragma unroll
    for (int k = 0; k < 11; ++k) Alds[wave][lane][k] = (_Float16)((inp[k] - mu) * inv);
    #pragma unroll
    for (int k = 11; k < 32; ++k) Alds[wave][lane][k] = (_Float16)0.0f;
    __syncthreads();

    // B fragments: lanes 0-15 carry K=0..15, lanes 16-31 K=16..31; col = lane&15
    v16h bfrag[2];
    const int kB = hs ? 16 : 0;
    #pragma unroll
    for (int u = 0; u < 2; ++u)
        #pragma unroll
        for (int i = 0; i < 16; ++i)
            bfrag[u][i] = W1s[kB + i][u * 16 + lrow];

    // A fragments: lanes 0-15 K=0..15, lanes 16-31 K=16..31; row = lane&15
    const int kA = hs ? 16 : 0;
    v8f acc[2][2];
    #pragma unroll
    for (int t = 0; t < 2; ++t) {
        v16h a;
        #pragma unroll
        for (int i = 0; i < 8; ++i) a[i]     = Alds[wave][t * 16 + lrow][kA + i];
        #pragma unroll
        for (int i = 0; i < 8; ++i) a[8 + i] = Alds[wave][t * 16 + lrow][kA + 8 + i];
        #pragma unroll
        for (int u = 0; u < 2; ++u) {
            v8f c = {};
            acc[t][u] = __builtin_amdgcn_wmma_f32_16x16x32_f16(
                false, a, false, bfrag[u], (short)0, c, false, false);
        }
    }

    // Bias + ReLU, scatter C layout (VGPR v -> M = v + 8*hs, N = lane&15) to LDS
    #pragma unroll
    for (int t = 0; t < 2; ++t)
        #pragma unroll
        for (int u = 0; u < 2; ++u) {
            int col = u * 16 + lrow;
            float bb = b1[col];
            #pragma unroll
            for (int v = 0; v < 8; ++v) {
                int row = t * 16 + v + hs * 8;
                Hlds[wave][row][col] = fmaxf(acc[t][u][v] + bb, 0.0f);
            }
        }
    __syncthreads();

    // Layer 2: per-lane dot over the lane's node row, stable softplus
    float z = b2[0];
    #pragma unroll
    for (int k = 0; k < 32; ++k) z += Hlds[wave][lane][k] * W2[k];
    float sp = fmaxf(z, 0.0f) + log1pf(expf(-fabsf(z)));
    if (node < N) atomicAdd(alpha_sum, sp);
}

// ---------------------------------------------------------------------------
// Kernel 3: per-node x-update — 2x2 solve via Cramer (D=2), alpha = mean.
// ---------------------------------------------------------------------------
__global__ void x_solve_kernel(const float* __restrict__ x,
                               const float* __restrict__ y,
                               const float* __restrict__ lam,
                               const float* __restrict__ bi,
                               const float* __restrict__ B,   // [2,2] row-major
                               const float* __restrict__ lam_agg,
                               const float* __restrict__ y_agg,
                               const float* __restrict__ w_deg,
                               const float* __restrict__ deg_sq,
                               const float* __restrict__ alpha_sum,
                               float* __restrict__ out_x,
                               long N) {
    long i = (long)blockIdx.x * blockDim.x + threadIdx.x;
    if (i >= N) return;
    const float alpha = alpha_sum[0] * (1.0f / (float)N);
    const float B00 = B[0], B01 = B[1], B10 = B[2], B11 = B[3];
    // 2 * (B^T B)
    const float t00 = 2.0f * (B00 * B00 + B10 * B10);
    const float t01 = 2.0f * (B00 * B01 + B10 * B11);
    const float t11 = 2.0f * (B01 * B01 + B11 * B11);

    const float wd  = w_deg[i];
    const float mii = wd * wd + deg_sq[i];
    const float bi0 = bi[2*i], bi1 = bi[2*i + 1];

    const float r0 = 2.0f * (bi0 * B00 + bi1 * B10)
                   - lam_agg[2*i]     - wd * lam[2*i]
                   + alpha * (mii * x[2*i]     - y_agg[2*i]     - wd * y[2*i]);
    const float r1 = 2.0f * (bi0 * B01 + bi1 * B11)
                   - lam_agg[2*i + 1] - wd * lam[2*i + 1]
                   + alpha * (mii * x[2*i + 1] - y_agg[2*i + 1] - wd * y[2*i + 1]);

    const float a00 = t00 + alpha * mii;
    const float a11 = t11 + alpha * mii;
    const float det = a00 * a11 - t01 * t01;
    const float id  = 1.0f / det;
    out_x[2*i]     = (r0 * a11 - t01 * r1) * id;
    out_x[2*i + 1] = (a00 * r1 - r0 * t01) * id;
}

// ---------------------------------------------------------------------------
// Kernel 4: edge pass 2 — scatter -w * new_x[sender] and in-degree count.
// ---------------------------------------------------------------------------
__global__ void edge_pass2_kernel(const float* __restrict__ edges,
                                  const int*   __restrict__ senders,
                                  const int*   __restrict__ receivers,
                                  const float* __restrict__ new_x,
                                  float* __restrict__ x_agg,
                                  float* __restrict__ deg,
                                  long E) {
    long stride = (long)gridDim.x * blockDim.x;
    for (long e = (long)blockIdx.x * blockDim.x + threadIdx.x; e < E; e += stride) {
        if (e + stride < E) {
            __builtin_prefetch(&edges[e + stride], 0, 0);
            __builtin_prefetch(&senders[e + stride], 0, 0);
            __builtin_prefetch(&receivers[e + stride], 0, 0);
        }
        float w = edges[e];
        int s = senders[e];
        int r = receivers[e];
        float nx0 = new_x[2*s], nx1 = new_x[2*s + 1];
        atomicAdd(&x_agg[2*r],     -w * nx0);
        atomicAdd(&x_agg[2*r + 1], -w * nx1);
        atomicAdd(&deg[r], 1.0f);
    }
}

// ---------------------------------------------------------------------------
// Kernel 5: per-node y / lambda update.
// ---------------------------------------------------------------------------
__global__ void y_lam_kernel(const float* __restrict__ lam,
                             const float* __restrict__ new_x,
                             const float* __restrict__ x_agg,
                             const float* __restrict__ w_deg,
                             const float* __restrict__ deg,
                             const float* __restrict__ alpha_sum,
                             float* __restrict__ out_y,
                             float* __restrict__ out_l,
                             long N) {
    long i = (long)blockIdx.x * blockDim.x + threadIdx.x;
    if (i >= N) return;
    const float alpha = alpha_sum[0] * (1.0f / (float)N);
    const float wd = w_deg[i];
    const float inv = 1.0f / (1.0f + deg[i]);
    const float ny0 = (x_agg[2*i]     + wd * new_x[2*i])     * inv;
    const float ny1 = (x_agg[2*i + 1] + wd * new_x[2*i + 1]) * inv;
    out_y[2*i]     = ny0;
    out_y[2*i + 1] = ny1;
    out_l[2*i]     = lam[2*i]     + alpha * ny0;
    out_l[2*i + 1] = lam[2*i + 1] + alpha * ny1;
}

// ---------------------------------------------------------------------------
extern "C" void kernel_launch(void* const* d_in, const int* in_sizes, int n_in,
                              void* d_out, int out_size, void* d_ws, size_t ws_size,
                              hipStream_t stream) {
    const float* x     = (const float*)d_in[0];
    const float* y     = (const float*)d_in[1];
    const float* lam   = (const float*)d_in[2];
    const float* bi    = (const float*)d_in[3];
    const float* edges = (const float*)d_in[4];
    const float* B     = (const float*)d_in[5];
    const float* W1    = (const float*)d_in[6];
    const float* b1    = (const float*)d_in[7];
    const float* W2    = (const float*)d_in[8];
    const float* b2    = (const float*)d_in[9];
    const int* senders   = (const int*)d_in[10];
    const int* receivers = (const int*)d_in[11];

    const long N = (long)in_sizes[0] / 2;   // x is [N, 2]
    const long E = (long)in_sizes[4];       // edges is [E, 1]

    // Workspace layout (floats): lam_agg[2N] y_agg[2N] w_deg[N] deg_sq[N]
    //                            x_agg[2N] deg[N] alpha_sum[1]
    float* ws        = (float*)d_ws;
    float* lam_agg   = ws;
    float* y_agg     = ws + 2 * N;
    float* w_deg     = ws + 4 * N;
    float* deg_sq    = ws + 5 * N;
    float* x_agg     = ws + 6 * N;
    float* deg       = ws + 8 * N;
    float* alpha_sum = ws + 9 * N;

    float* out   = (float*)d_out;   // [3, N, 2] flat
    float* out_x = out;
    float* out_y = out + 2 * N;
    float* out_l = out + 4 * N;

    const long zn = 9 * N + 1;
    zero_ws_kernel<<<(int)((zn + 255) / 256), 256, 0, stream>>>(ws, zn);

    const int EBLOCKS = 4096;  // grid-stride: ~8 edges per thread at E=8M
    edge_pass1_kernel<<<EBLOCKS, 256, 0, stream>>>(edges, senders, receivers,
                                                   lam, y, lam_agg, y_agg,
                                                   w_deg, deg_sq, E);

    node_mlp_alpha_kernel<<<(int)((N + 127) / 128), 128, 0, stream>>>(
        x, y, lam, lam_agg, y_agg, w_deg, W1, b1, W2, b2, alpha_sum, N);

    x_solve_kernel<<<(int)((N + 255) / 256), 256, 0, stream>>>(
        x, y, lam, bi, B, lam_agg, y_agg, w_deg, deg_sq, alpha_sum, out_x, N);

    edge_pass2_kernel<<<EBLOCKS, 256, 0, stream>>>(edges, senders, receivers,
                                                   out_x, x_agg, deg, E);

    y_lam_kernel<<<(int)((N + 255) / 256), 256, 0, stream>>>(
        lam, out_x, x_agg, w_deg, deg, alpha_sum, out_y, out_l, N);
}
